// TFNet_71571335020819
// MI455X (gfx1250) — matrix operand order repaired and checked by reading, not compiled, not run on hardware
//
#include <hip/hip_runtime.h>

// ---------------------------------------------------------------------------
// TFNet forward for MI455X (gfx1250, wave32, WMMA f32<=f16 16x16x32).
//
// - emb folded into conv weights per batch -> all heavy ops are K>=32 GEMMs.
// - dna im2col window loaded via TDM (tensor_load_to_lds) + s_wait_tensorcnt.
// - LDS staging buffers stored pre-swizzled in WMMA B-fragment order so each
//   fragment is one aligned 32B read (2x ds_load_b128) per lane.
// Workspace: emb 15KB + W2 28.1MB + xp 108.9MB + svec 0.57MB + h 32KB.
// ---------------------------------------------------------------------------

typedef __attribute__((ext_vector_type(16))) _Float16     v16h;
typedef __attribute__((ext_vector_type(8)))  _Float16     v8h;
typedef __attribute__((ext_vector_type(8)))  float        v8f;
typedef __attribute__((ext_vector_type(4)))  unsigned int u32x4;
typedef __attribute__((ext_vector_type(8)))  int          i32x8;
typedef __attribute__((ext_vector_type(4)))  int          i32x4;

#define NB    16
#define LSEQ  1000
#define TT    39
#define EE    6
#define SOUT  984
#define NTILE 62
#define CTOT  384
#define TP    9

__device__ __forceinline__ int akmap(int h, int g) {      // A frag: f16 16x32
    return h + 16 * g;
}

__device__ __forceinline__ v8f wmma32f16(v16h a, v16h b, v8f c) {
    return __builtin_amdgcn_wmma_f32_16x16x32_f16(false, a, false, b,
                                                  (short)0, c, false, false);
}

#define V8F0 (v8f){0.f,0.f,0.f,0.f,0.f,0.f,0.f,0.f}

// Swizzled B staging: element (K in [0,32), n in [0,16)) of chunk kc lives at
// [kc*512 + ((K&16)+n)*16 + (K&15)]; lane l then reads halves [l*16 .. l*16+15].
__device__ __forceinline__ int bswz(int K, int n) {
    return ((K & 16) + n) * 16 + (K & 15);
}

// ---------------------------------------------------------------------------
// TDM: 1-D tile of fp32 from global -> LDS. D# per CDNA5 ISA 8.3/8.4.
// tensor_dim0 = elements remaining in the tensor => OOB tail reads return 0.
// 6-arg builtin form (clang-23 / therock HIP headers).
// ---------------------------------------------------------------------------
__device__ __forceinline__ void tdm_load_1d_f32(unsigned lds_off, const float* gaddr,
                                                unsigned nelem, unsigned tensor_len) {
    unsigned long long ga = (unsigned long long)(uintptr_t)gaddr;
    u32x4 g0;
    g0[0] = 1u;                                            // count=1 (valid), user mode
    g0[1] = lds_off;                                       // LDS byte address
    g0[2] = (unsigned)(ga & 0xffffffffu);                  // global_addr[31:0]
    g0[3] = (unsigned)((ga >> 32) & 0x01ffffffu)           // global_addr[56:32]
          | (2u << 30);                                    // type=2 ("image")
    i32x8 g1;
    g1[0] = (int)(2u << 16);                               // data_size=2 (4B)
    g1[1] = (int)((tensor_len & 0xffffu) << 16);           // tensor_dim0[15:0]
    g1[2] = (int)((tensor_len >> 16) & 0xffffu)            // tensor_dim0[31:16]
          | (1 << 16);                                     // tensor_dim1 = 1
    g1[3] = (int)(nelem << 16);                            // tile_dim0
    g1[4] = 0;                                             // tile_dim1/2 unused
    g1[5] = (int)tensor_len;                               // dim0 stride (unused, 1D)
    g1[6] = 0;
    g1[7] = 0;
    i32x4 z4 = {0, 0, 0, 0};
    i32x8 z8 = {0, 0, 0, 0, 0, 0, 0, 0};
    __builtin_amdgcn_tensor_load_to_lds(g0, g1, z4, z4, z8, 0);
}

// ---------------------------------------------------------------------------
// 1) emb[b,j,:] = table[tf[b,j],:]
// ---------------------------------------------------------------------------
__global__ void k_embed(const float* __restrict__ table, const int* __restrict__ tf,
                        float* __restrict__ emb) {
    int idx = blockIdx.x * blockDim.x + threadIdx.x;
    if (idx >= NB * TT) return;
    int v = tf[idx];
    #pragma unroll
    for (int e = 0; e < EE; ++e) emb[idx * EE + e] = table[v * EE + e];
}

// ---------------------------------------------------------------------------
// 2) W2[b,c,j,ka] = sum_e w[c,k,a,e]*emb[b,j,e]   (fp16, zero-padded K)
// ---------------------------------------------------------------------------
__global__ void k_combine(const float* __restrict__ w, const float* __restrict__ emb,
                          _Float16* __restrict__ W2, int C, int cs, int Kp, long N) {
    long idx = (long)blockIdx.x * blockDim.x + threadIdx.x;
    if (idx >= N) return;
    int ka = (int)(idx % Kp); long t = idx / Kp;
    int j = (int)(t % TT);    t /= TT;
    int c = (int)(t % C);     int b = (int)(t / C);
    float s = 0.f;
    if (ka < cs * 4) {
        int k = ka >> 2, a = ka & 3;
        const float* wp = w + ((size_t)(c * cs + k) * 4 + a) * EE;
        const float* ep = emb + ((size_t)b * TT + j) * EE;
        #pragma unroll
        for (int e = 0; e < EE; ++e) s += wp[e] * ep[e];
    }
    W2[idx] = (_Float16)s;
}

// ---------------------------------------------------------------------------
// 3) Interaction GEMM + bias+relu+BN+relu + maxpool(T,4) -> xp fp16
//    dna window staged via TDM; B frags built once, reused over all 36 j's.
// ---------------------------------------------------------------------------
template <int CS, int OFF, int KP, int CBR, int CBASE>
__global__ void __launch_bounds__(128)
k_interact(const _Float16* __restrict__ W2, const float* __restrict__ dna,
           const float* __restrict__ bconv, const float* __restrict__ gbn,
           const float* __restrict__ bebn, _Float16* __restrict__ xp) {
    constexpr int NCH  = KP / 32;
    constexpr int NPOS = 15 + CS;                   // dna positions needed
    __shared__ float dwin[NPOS * 4];
    const int lane = threadIdx.x & 31;
    const int wv   = threadIdx.x >> 5;
    const int g    = lane >> 4;
    const int col  = lane & 15;
    int bi = blockIdx.x;
    const int it = bi % NTILE;  bi /= NTILE;
    const int ct = bi % (CBR / 16);
    const int bb = bi / (CBR / 16);
    const int i  = it * 16 + col;
    const float rs = rsqrtf(1.0f + 1e-5f);

    // TDM DMA: dna[bb, it*16+OFF .. +NPOS, 0..3] -> dwin (clipped at dna end).
    if (wv == 0) {
        const int p0 = it * 16 + OFF;
        tdm_load_1d_f32((unsigned)(uintptr_t)&dwin[0],
                        dna + ((size_t)bb * LSEQ + p0) * 4,
                        NPOS * 4, (unsigned)((LSEQ - p0) * 4));
        __builtin_amdgcn_s_wait_tensorcnt(0);
    }
    __syncthreads();

    // B fragments from LDS: B[K=(k,a), n=col] = dwin[(col+k)*4 + a].
    v16h bf[NCH];
    #pragma unroll
    for (int kc = 0; kc < NCH; ++kc) {
        #pragma unroll
        for (int h = 0; h < 16; ++h) {
            int K = kc * 32 + h + 16 * g;
            int Kc = (K < CS * 4) ? K : 0;          // clamp into dwin
            float v = dwin[(col + (Kc >> 2)) * 4 + (Kc & 3)];
            bf[kc][h] = (_Float16)((K < CS * 4) ? v : 0.f);
        }
    }

    float bc[8], sc[8], bt[8];
    #pragma unroll
    for (int v = 0; v < 8; ++v) {
        int cl = ct * 16 + v + 8 * g;
        bc[v] = bconv[cl];
        sc[v] = gbn[cl] * rs;
        bt[v] = bebn[cl];
    }

    const int crow = ct * 16 + col;                 // A row (channel)
    for (int t = wv; t < TP; t += 4) {
        float pm[8] = {0,0,0,0,0,0,0,0};
        for (int jj = 0; jj < 4; ++jj) {
            int j = t * 4 + jj;
            v8f acc = V8F0;
            size_t abase = (((size_t)bb * CBR + crow) * TT + j) * KP;
            #pragma unroll
            for (int kc = 0; kc < NCH; ++kc) {
                v16h af;
                #pragma unroll
                for (int h = 0; h < 16; ++h)
                    af[h] = W2[abase + kc * 32 + akmap(h, g)];
                acc = wmma32f16(af, bf[kc], acc);
            }
            #pragma unroll
            for (int v = 0; v < 8; ++v) {
                float y = fmaxf(acc[v] + bc[v], 0.f);   // bias + relu
                y = fmaxf(y * sc[v] + bt[v], 0.f);      // BN + relu
                pm[v] = fmaxf(pm[v], y);                // maxpool over j
            }
        }
        if (i < SOUT) {
            #pragma unroll
            for (int v = 0; v < 8; ++v) {
                int cg = CBASE + ct * 16 + v + 8 * g;
                xp[(((size_t)bb * CTOT + cg) * TP + t) * SOUT + i] = (_Float16)pm[v];
            }
        }
    }
}

// ---------------------------------------------------------------------------
// 4) Fused lin0(256x384)+BN+ReLU -> lin1(128x256)+BN+ReLU -> single(1x128)
//    LDS activations pre-swizzled in fragment order -> 32B B-frag reads.
// ---------------------------------------------------------------------------
__global__ void __launch_bounds__(256)
k_lin(const _Float16* __restrict__ xp,
      const float* __restrict__ w0, const float* __restrict__ b0,
      const float* __restrict__ g0, const float* __restrict__ e0,
      const float* __restrict__ w1, const float* __restrict__ b1,
      const float* __restrict__ g1, const float* __restrict__ e1,
      const float* __restrict__ wS, const float* __restrict__ bS,
      float* __restrict__ svec) {
    __shared__ __align__(32) _Float16 xsw[12 * 512];    // 384-ch input, swizzled
    __shared__ __align__(32) _Float16 z0sw[8 * 512];    // 256-ch, swizzled
    __shared__ __align__(32) _Float16 z1sw[4 * 512];    // 128-ch, swizzled
    const int tid  = threadIdx.x;
    const int lane = tid & 31, wv = tid >> 5, g = lane >> 4, col = lane & 15;
    int bi = blockIdx.x;
    const int it = bi % NTILE;  bi /= NTILE;
    const int t  = bi % TP;
    const int bb = bi / TP;
    const int i0 = it * 16;
    const float rs = rsqrtf(1.0f + 1e-5f);

    // Stage xp rows (32B contiguous) into swizzled LDS.
    for (int ch = tid; ch < CTOT; ch += 256) {
        const _Float16* row = &xp[(((size_t)bb * CTOT + ch) * TP + t) * SOUT + i0];
        v8h r0 = *(const v8h*)row;
        v8h r1 = *(const v8h*)(row + 8);
        int kc = ch >> 5, K = ch & 31;
        #pragma unroll
        for (int ii = 0; ii < 16; ++ii) {
            _Float16 v = (ii < 8) ? r0[ii] : r1[ii - 8];
            if (i0 + ii >= SOUT) v = (_Float16)0.f;
            xsw[kc * 512 + bswz(K, ii)] = v;
        }
    }
    __syncthreads();

    // lin0: [256 x 384] @ [384 x 16]; each wave owns 2 M-tiles.
    #pragma unroll
    for (int mi = 0; mi < 2; ++mi) {
        const int m = wv * 2 + mi;
        const int o = m * 16 + col;
        v8f acc = V8F0;
        #pragma unroll
        for (int kc = 0; kc < 12; ++kc) {
            v16h af;
            #pragma unroll
            for (int h = 0; h < 16; ++h)
                af[h] = (_Float16)w0[(size_t)o * 384 + kc * 32 + akmap(h, g)];
            v16h bfm = *(const v16h*)&xsw[kc * 512 + lane * 16];
            acc = wmma32f16(af, bfm, acc);
        }
        #pragma unroll
        for (int v = 0; v < 8; ++v) {
            int oo = m * 16 + v + 8 * g;
            float y = acc[v] + b0[oo];
            y = fmaxf(y * (g0[oo] * rs) + e0[oo], 0.f);
            z0sw[(oo >> 5) * 512 + bswz(oo & 31, col)] = (_Float16)y;
        }
    }
    __syncthreads();

    // lin1: [128 x 256] @ [256 x 16]; one M-tile per wave.
    {
        const int o = wv * 16 + col;
        v8f acc = V8F0;
        #pragma unroll
        for (int kc = 0; kc < 8; ++kc) {
            v16h af;
            #pragma unroll
            for (int h = 0; h < 16; ++h)
                af[h] = (_Float16)w1[(size_t)o * 256 + kc * 32 + akmap(h, g)];
            v16h bfm = *(const v16h*)&z0sw[kc * 512 + lane * 16];
            acc = wmma32f16(af, bfm, acc);
        }
        #pragma unroll
        for (int v = 0; v < 8; ++v) {
            int oo = wv * 16 + v + 8 * g;
            float y = acc[v] + b1[oo];
            y = fmaxf(y * (g1[oo] * rs) + e1[oo], 0.f);
            z1sw[(oo >> 5) * 512 + bswz(oo & 31, col)] = (_Float16)y;
        }
    }
    __syncthreads();

    // single: 1x128 head, 16 output columns.
    if (tid < 16 && i0 + tid < SOUT) {
        float s = bS[0];
        #pragma unroll 8
        for (int ch = 0; ch < 128; ++ch)
            s += wS[ch] * (float)z1sw[(ch >> 5) * 512 + bswz(ch & 31, tid)];
        svec[(size_t)bb * (TP * SOUT) + t * SOUT + i0 + tid] = s;
    }
}

// ---------------------------------------------------------------------------
// 5) full0: h = relu(svec[16,8856] @ w_full0^T + b). M=16 = batch dimension.
// ---------------------------------------------------------------------------
__global__ void __launch_bounds__(256)
k_head(const float* __restrict__ svec, const float* __restrict__ wf0,
       const float* __restrict__ bf0, float* __restrict__ hbuf) {
    const int tid  = threadIdx.x;
    const int lane = tid & 31, wv = tid >> 5, g = lane >> 4, col = lane & 15;
    const int KD = TP * SOUT;                        // 8856
    const int KCH = (KD + 31) / 32;                  // 277
    #pragma unroll
    for (int nti = 0; nti < 4; ++nti) {
        const int n = (wv * 4 + nti) * 16 + col;
        v8f acc = V8F0;
        for (int kc = 0; kc < KCH; ++kc) {
            if (kc + 1 < KCH)                        // global_prefetch_b8
                __builtin_prefetch(&wf0[(size_t)n * KD + (kc + 1) * 32], 0, 0);
            v16h af, bfm;
            #pragma unroll
            for (int h = 0; h < 16; ++h) {
                int K = kc * 32 + akmap(h, g);
                af[h] = (_Float16)((K < KD) ? svec[(size_t)col * KD + K] : 0.f);
            }
            #pragma unroll
            for (int h = 0; h < 16; ++h) {
                int K = kc * 32 + h + 16 * g;
                bfm[h] = (_Float16)((K < KD) ? wf0[(size_t)n * KD + K] : 0.f);
            }
            acc = wmma32f16(af, bfm, acc);
        }
        #pragma unroll
        for (int v = 0; v < 8; ++v) {
            int brow = v + 8 * g;
            hbuf[(size_t)brow * 512 + n] = fmaxf(acc[v] + bf0[n], 0.f);
        }
    }
}

// ---------------------------------------------------------------------------
// 6) full1: out[b] = h[b,:].w_full1 + b_full1  (deterministic tree reduce)
// ---------------------------------------------------------------------------
__global__ void __launch_bounds__(256)
k_final(const float* __restrict__ hbuf, const float* __restrict__ wf1,
        const float* __restrict__ bf1, float* __restrict__ out) {
    __shared__ float red[256];
    const int tid = threadIdx.x;
    for (int b = 0; b < NB; ++b) {
        float s = 0.f;
        for (int t = tid; t < 512; t += 256) s += wf1[t] * hbuf[b * 512 + t];
        red[tid] = s;
        __syncthreads();
        for (int st = 128; st > 0; st >>= 1) {
            if (tid < st) red[tid] += red[tid + st];
            __syncthreads();
        }
        if (tid == 0) out[b] = red[0] + bf1[0];
        __syncthreads();
    }
}

// ---------------------------------------------------------------------------
extern "C" void kernel_launch(void* const* d_in, const int* in_sizes, int n_in,
                              void* d_out, int out_size, void* d_ws, size_t ws_size,
                              hipStream_t stream) {
    (void)in_sizes; (void)n_in; (void)out_size; (void)ws_size;
    const float* DNA   = (const float*)d_in[0];
    const int*   tf    = (const int*)  d_in[1];
    const float* table = (const float*)d_in[2];
    const float *wconv[4], *bconv[4], *gbn[4], *bebn[4];
    for (int i = 0; i < 4; ++i) {
        wconv[i] = (const float*)d_in[3 + 4 * i];
        bconv[i] = (const float*)d_in[4 + 4 * i];
        gbn[i]   = (const float*)d_in[5 + 4 * i];
        bebn[i]  = (const float*)d_in[6 + 4 * i];
    }
    const float* w0 = (const float*)d_in[19]; const float* b0 = (const float*)d_in[20];
    const float* g0 = (const float*)d_in[21]; const float* e0 = (const float*)d_in[22];
    const float* w1 = (const float*)d_in[23]; const float* b1 = (const float*)d_in[24];
    const float* g1 = (const float*)d_in[25]; const float* e1 = (const float*)d_in[26];
    const float* wS = (const float*)d_in[27]; const float* bS = (const float*)d_in[28];
    const float* wf0 = (const float*)d_in[29]; const float* bf0 = (const float*)d_in[30];
    const float* wf1 = (const float*)d_in[31]; const float* bf1 = (const float*)d_in[32];
    float* out = (float*)d_out;

    static const int C[4]  = {128, 128, 64, 64};
    static const int CS[4] = {5, 9, 13, 17};
    static const int KP[4] = {32, 64, 64, 96};

    char* p = (char*)d_ws;
    auto carve = [&](size_t bytes) -> char* {
        char* r = p; p += (bytes + 255) & ~(size_t)255; return r;
    };
    float*    emb = (float*)carve((size_t)NB * TT * EE * sizeof(float));
    _Float16* W2[4];
    for (int i = 0; i < 4; ++i)
        W2[i] = (_Float16*)carve((size_t)NB * C[i] * TT * KP[i] * sizeof(_Float16));
    _Float16* xp   = (_Float16*)carve((size_t)NB * CTOT * TP * SOUT * sizeof(_Float16) + 256);
    float*    svec = (float*)carve((size_t)NB * TP * SOUT * sizeof(float));
    float*    hbuf = (float*)carve((size_t)NB * 512 * sizeof(float));

    k_embed<<<(NB * TT + 127) / 128, 128, 0, stream>>>(table, tf, emb);

    for (int i = 0; i < 4; ++i) {
        long N = (long)NB * C[i] * TT * KP[i];
        k_combine<<<(int)((N + 255) / 256), 256, 0, stream>>>(
            wconv[i], emb, W2[i], C[i], CS[i], KP[i], N);
    }

    k_interact<5, 6, 32, 128, 0><<<NB * (128 / 16) * NTILE, 128, 0, stream>>>(
        W2[0], DNA, bconv[0], gbn[0], bebn[0], xp);
    k_interact<9, 4, 64, 128, 128><<<NB * (128 / 16) * NTILE, 128, 0, stream>>>(
        W2[1], DNA, bconv[1], gbn[1], bebn[1], xp);
    k_interact<13, 2, 64, 64, 256><<<NB * (64 / 16) * NTILE, 128, 0, stream>>>(
        W2[2], DNA, bconv[2], gbn[2], bebn[2], xp);
    k_interact<17, 0, 96, 64, 320><<<NB * (64 / 16) * NTILE, 128, 0, stream>>>(
        W2[3], DNA, bconv[3], gbn[3], bebn[3], xp);

    k_lin<<<NB * TP * NTILE, 256, 0, stream>>>(
        xp, w0, b0, g0, e0, w1, b1, g1, e1, wS, bS, svec);

    k_head<<<1, 256, 0, stream>>>(svec, wf0, bf0, hbuf);
    k_final<<<1, 256, 0, stream>>>(hbuf, wf1, bf1, out);
}
